// LSTMDecoder_15891378995820
// MI455X (gfx1250) — compile-verified, matrix operand order checked by reading, not run on hardware
//
#include <hip/hip_runtime.h>
#include <hip/hip_bf16.h>

// ---------------------------------------------------------------------------
// LSTM decoder: B=4096, LATENT=128, H=256, OUT=88, T=64, L=5
// Persistent WMMA-bf16 LSTM kernel, 16 batch rows / workgroup,
// h in LDS (bf16), c in registers, weights pre-packed to bf16 fragments.
// - Opaque per-timestep pointers stop LICM from hoisting fragment loads.
// - Explicit addrspace(1) loads keep all streams on GLOBAL (LOADcnt only).
// - gx0 pre-packed into D-fragment layout (b128 loads); biases pre-summed.
// ---------------------------------------------------------------------------

#define B_SZ     4096
#define LATENT   128
#define HID      256
#define OUTN     88
#define T_STEPS  64
#define NLAYERS  5
#define GATES    1024   // 4*HID

typedef __attribute__((ext_vector_type(16))) __bf16        v16bf;
typedef __attribute__((ext_vector_type(8)))  float         v8f;
typedef __attribute__((ext_vector_type(4)))  unsigned int  v4u;   // plain vector, not HIP class

union FragB16 {
    v16bf v;
    unsigned int u[8];
    v4u q[2];
};
union Frag32 {
    v8f v;
    v4u q[2];
};

// Force-global loads (survive opaque/generic pointers).
__device__ __forceinline__ v4u gld_b128(const void* p) {
    return *(const __attribute__((address_space(1))) v4u*)(uintptr_t)p;
}
__device__ __forceinline__ float gld_f32(const float* p) {
    return *(const __attribute__((address_space(1))) float*)(uintptr_t)p;
}

// workspace layout (bytes)
static constexpr size_t OFF_GX0  = 0;                                   // B*1024 f32 (fragment-major)
static constexpr size_t OFF_XP   = OFF_GX0 + (size_t)B_SZ * GATES * 4;  // B*256 f32
static constexpr size_t OFF_WHH  = OFF_XP  + (size_t)B_SZ * HID * 4;    // 5 * 262144 bf16
static constexpr size_t OFF_WIH  = OFF_WHH + (size_t)5 * 262144 * 2;    // 4 * 262144 bf16 (layers 1..4)
static constexpr size_t OFF_OUTW = OFF_WIH + (size_t)4 * 262144 * 2;    // 6*8*512 bf16
static constexpr size_t OFF_BSUM = OFF_OUTW + (size_t)6 * 8 * 512 * 2;  // 5*1024 f32

// ---------------------------------------------------------------------------
// Phase 0a: xp = x @ linear_w.T + linear_b               [B, 256]
// ---------------------------------------------------------------------------
__global__ void k_xp(const float* __restrict__ x, const float* __restrict__ lw,
                     const float* __restrict__ lb, float* __restrict__ xp) {
    int idx = blockIdx.x * blockDim.x + threadIdx.x;
    if (idx >= B_SZ * HID) return;
    int b = idx >> 8, hh = idx & 255;
    const float* xr = x + (size_t)b * LATENT;
    const float* wr = lw + (size_t)hh * LATENT;
    float acc = lb[hh];
#pragma unroll 4
    for (int k = 0; k < LATENT; ++k) acc = fmaf(xr[k], wr[k], acc);
    xp[idx] = acc;
}

// ---------------------------------------------------------------------------
// Phase 0b: gx0 = xp @ w_ih[0].T + b_ih[0] + b_hh[0], written directly in
// D-fragment-major layout: gx0p[((block*64 + ntile)*32 + lane)*8 + r]
//   block = b>>4, r = b&7, half = (b>>3)&1, lane = half*16 + (n&15),
//   ntile = n>>4   -> each lane's 8 accumulator values are 32B contiguous.
// ---------------------------------------------------------------------------
__global__ void k_gx0(const float* __restrict__ xp, const float* __restrict__ wih0,
                      const float* __restrict__ bih0, const float* __restrict__ bhh0,
                      float* __restrict__ gx0p) {
    int idx = blockIdx.x * blockDim.x + threadIdx.x;
    if (idx >= B_SZ * GATES) return;
    int b = idx >> 10, n = idx & 1023;
    const float* xr = xp + (size_t)b * HID;
    const float* wr = wih0 + (size_t)n * HID;
    float acc = bih0[n] + bhh0[n];
#pragma unroll 4
    for (int k = 0; k < HID; ++k) acc = fmaf(xr[k], wr[k], acc);

    int block = b >> 4;
    int r     = b & 7;
    int half  = (b >> 3) & 1;
    int ntile = n >> 4;
    int lane  = half * 16 + (n & 15);
    gx0p[(((size_t)block * 64 + ntile) * 32 + lane) * 8 + r] = acc;
}

// ---------------------------------------------------------------------------
// Phase 0b': pre-summed gate biases: bsum = b_ih + b_hh  [5*1024]
// ---------------------------------------------------------------------------
__global__ void k_bsum(const float* __restrict__ bih, const float* __restrict__ bhh,
                       float* __restrict__ bsum) {
    int idx = blockIdx.x * blockDim.x + threadIdx.x;
    if (idx >= NLAYERS * GATES) return;
    bsum[idx] = bih[idx] + bhh[idx];
}

// ---------------------------------------------------------------------------
// Phase 0c: pack W[n_rows x 256] (f32, row-major over K) into bf16 WMMA
// B-fragments: dst[frag][lane][j], frag = ntile*8 + ktile, 512 bf16 per frag.
//   n = ntile*16 + (lane&15)
//   k = ktile*32 + (p<4 ? 2p : 16+2(p-4)) + (j&1) + (lane>=16 ? 8 : 0),  p=j>>1
// ---------------------------------------------------------------------------
__global__ void k_pack(const float* __restrict__ W, __bf16* __restrict__ dst,
                       int n_rows, int n_elems) {
    int idx = blockIdx.x * blockDim.x + threadIdx.x;
    if (idx >= n_elems) return;
    int frag = idx >> 9;
    int e    = idx & 511;
    int lane = e >> 4, j = e & 15;
    int ntile = frag >> 3, kt = frag & 7;
    int n = ntile * 16 + (lane & 15);
    int p = j >> 1, wbit = j & 1;
    int k = kt * 32 + ((p < 4) ? p * 2 : 16 + (p - 4) * 2) + wbit + ((lane >= 16) ? 8 : 0);
    float v = (n < n_rows) ? W[(size_t)n * HID + k] : 0.0f;
    dst[idx] = (__bf16)v;
}

// ---------------------------------------------------------------------------
// 8x8-fragment GEMM: acc[8] (+)= A(h LDS) x B(packed weights).
// Per K-chunk: 1 A-frag (LDS) + 8 staged B-frags (one global clause), then
// 8 back-to-back WMMAs.
// ---------------------------------------------------------------------------
#define GEMM8x8(LSRC, WBASE)                                                    \
    {                                                                           \
        _Pragma("unroll")                                                       \
        for (int kt = 0; kt < 8; ++kt) {                                        \
            FragB16 afr;                                                        \
            _Pragma("unroll")                                                   \
            for (int p = 0; p < 8; ++p) {                                       \
                int koff = kt * 32 + ((p < 4) ? p * 2 : 16 + (p - 4) * 2) + akhalf; \
                afr.u[p] = *(const unsigned int*)&(LSRC)[arow][koff];           \
            }                                                                   \
            FragB16 bfr[8];                                                     \
            _Pragma("unroll")                                                   \
            for (int tt = 0; tt < 8; ++tt) {                                    \
                int ntile = (tt >> 1) * 16 + wv * 2 + (tt & 1);                 \
                const __bf16* wp = (WBASE) + ((size_t)(ntile * 8 + kt) << 9) + lane * 16; \
                bfr[tt].q[0] = gld_b128(wp);                                    \
                bfr[tt].q[1] = gld_b128(wp + 8);                                \
            }                                                                   \
            _Pragma("unroll")                                                   \
            for (int tt = 0; tt < 8; ++tt)                                      \
                acc[tt] = __builtin_amdgcn_wmma_f32_16x16x32_bf16(              \
                    false, afr.v, false, bfr[tt].v, (short)0, acc[tt], false, false); \
        }                                                                       \
    }

__launch_bounds__(256, 1)
__global__ void k_lstm(const float* __restrict__ gx0p,   // fragment-major [B/16][64][32][8]
                       const __bf16* __restrict__ whh,   // [5][64*8*512]
                       const __bf16* __restrict__ wih,   // [4][64*8*512] (layers 1..4)
                       const __bf16* __restrict__ outw,  // [6*8*512], rows>=88 zero
                       const float* __restrict__ bsum,   // [5][1024] (b_ih+b_hh)
                       const float* __restrict__ out_b,  // [88]
                       float* __restrict__ out)          // [B][T][88]
{
    __shared__ __bf16 h_lds[NLAYERS][16][HID];   // 40 KB

    const int tid    = threadIdx.x;
    const int wv     = tid >> 5;              // wave id 0..7
    const int lane   = tid & 31;
    const int ncol   = lane & 15;             // B/C/D column within tile
    const int mhi8   = (lane >= 16) ? 8 : 0;  // C/D row offset
    const int arow   = lane & 15;             // A-fragment row
    const int akhalf = (lane >= 16) ? 8 : 0;  // A-fragment K offset
    const int mb     = blockIdx.x * 16;       // batch tile base

    // h0 = 0
    for (int i = tid; i < NLAYERS * 16 * HID; i += 256)
        ((__bf16*)h_lds)[i] = (__bf16)0.0f;
    __syncthreads();

    // c0 = 0 (registers: [layer][sub-tile][row])
    float cst[NLAYERS][2][8];
#pragma unroll
    for (int l = 0; l < NLAYERS; ++l)
#pragma unroll
        for (int s = 0; s < 2; ++s)
#pragma unroll
            for (int r = 0; r < 8; ++r) cst[l][s][r] = 0.0f;

#pragma unroll 1
    for (int t = 0; t < T_STEPS; ++t) {
        // Opaque per-iteration copies: block LICM from hoisting the fragment
        // loads across timesteps (address space re-imposed at load sites).
        const __bf16* whh_t  = whh;
        const __bf16* wih_t  = wih;
        const __bf16* outw_t = outw;
        const float*  gx0_t  = gx0p;
        const float*  bsum_t = bsum;
        asm volatile("" : "+s"(whh_t), "+s"(wih_t), "+s"(outw_t),
                          "+s"(gx0_t), "+s"(bsum_t));

#pragma unroll
        for (int l = 0; l < NLAYERS; ++l) {
            v8f acc[8];
            if (l == 0) {
                // accumulators = constant layer-0 input gates (fragment-major, b128s)
#pragma unroll
                for (int tt = 0; tt < 8; ++tt) {
                    int ntile = (tt >> 1) * 16 + wv * 2 + (tt & 1);
                    const float* gp =
                        gx0_t + (((size_t)blockIdx.x * 64 + ntile) * 32 + lane) * 8;
                    Frag32 f;
                    f.q[0] = gld_b128(gp);
                    f.q[1] = gld_b128(gp + 4);
                    acc[tt] = f.v;
                }
            } else {
#pragma unroll
                for (int tt = 0; tt < 8; ++tt) {
                    int n  = l * GATES + (tt >> 1) * HID + wv * 32 + (tt & 1) * 16 + ncol;
                    float bs = gld_f32(&bsum_t[n]);
#pragma unroll
                    for (int r = 0; r < 8; ++r) acc[tt][r] = bs;
                }
                // input GEMM: h_new[l-1] @ w_ih[l].T
                const __bf16* wib = wih_t + (size_t)(l - 1) * (64 * 8 * 512);
                GEMM8x8(h_lds[l - 1], wib)
            }
            // recurrent GEMM: h_prev[l] @ w_hh[l].T
            {
                const __bf16* whb = whh_t + (size_t)l * (64 * 8 * 512);
                GEMM8x8(h_lds[l], whb)
            }

            // elementwise LSTM cell (wave-local: gates i/f/g/o share hidden col)
            float hnew[2][8];
#pragma unroll
            for (int s = 0; s < 2; ++s)
#pragma unroll
                for (int r = 0; r < 8; ++r) {
                    float iv = acc[0 * 2 + s][r];
                    float fv = acc[1 * 2 + s][r];
                    float gv = acc[2 * 2 + s][r];
                    float ov = acc[3 * 2 + s][r];
                    float si = 1.0f / (1.0f + __expf(-iv));
                    float sf = 1.0f / (1.0f + __expf(-fv));
                    float so = 1.0f / (1.0f + __expf(-ov));
                    float tg = tanhf(gv);
                    float cn = sf * cst[l][s][r] + si * tg;
                    cst[l][s][r] = cn;
                    hnew[s][r] = so * tanhf(cn);
                }

            __syncthreads();   // all waves done reading h_lds[l] / h_lds[l-1]
#pragma unroll
            for (int s = 0; s < 2; ++s)
#pragma unroll
                for (int r = 0; r < 8; ++r)
                    h_lds[l][r + mhi8][wv * 32 + s * 16 + ncol] = (__bf16)hnew[s][r];
            __syncthreads();   // publish h_new[l]
        }

        // output projection: sigmoid(h[4] @ out_w.T + out_b), waves 0..5
        if (wv < 6) {
            int n = wv * 16 + ncol;
            float ob = (n < OUTN) ? out_b[n] : 0.0f;
            v8f acc2;
#pragma unroll
            for (int r = 0; r < 8; ++r) acc2[r] = ob;
#pragma unroll
            for (int kt = 0; kt < 8; ++kt) {
                FragB16 afr;
#pragma unroll
                for (int p = 0; p < 8; ++p) {
                    int koff = kt * 32 + ((p < 4) ? p * 2 : 16 + (p - 4) * 2) + akhalf;
                    afr.u[p] = *(const unsigned int*)&h_lds[4][arow][koff];
                }
                const __bf16* wp = outw_t + ((size_t)(wv * 8 + kt) << 9) + lane * 16;
                FragB16 bfr;
                bfr.q[0] = gld_b128(wp);
                bfr.q[1] = gld_b128(wp + 8);
                acc2 = __builtin_amdgcn_wmma_f32_16x16x32_bf16(
                    false, afr.v, false, bfr.v, (short)0, acc2, false, false);
            }
            if (n < OUTN) {
#pragma unroll
                for (int r = 0; r < 8; ++r) {
                    float v = 1.0f / (1.0f + __expf(-acc2[r]));
                    // stream out non-temporally: keep L2 for weights/gx0
                    __builtin_nontemporal_store(
                        v, &out[((size_t)(mb + r + mhi8) * T_STEPS + t) * OUTN + n]);
                }
            }
        }
        __syncthreads();
    }
}

// ---------------------------------------------------------------------------
extern "C" void kernel_launch(void* const* d_in, const int* in_sizes, int n_in,
                              void* d_out, int out_size, void* d_ws, size_t ws_size,
                              hipStream_t stream) {
    const float* x    = (const float*)d_in[0];
    const float* lw   = (const float*)d_in[1];
    const float* lb   = (const float*)d_in[2];
    const float* w_ih = (const float*)d_in[3];   // [5][1024][256]
    const float* w_hh = (const float*)d_in[4];   // [5][1024][256]
    const float* b_ih = (const float*)d_in[5];   // [5][1024]
    const float* b_hh = (const float*)d_in[6];   // [5][1024]
    const float* ow   = (const float*)d_in[7];   // [88][256]
    const float* ob   = (const float*)d_in[8];   // [88]
    float* out = (float*)d_out;

    char* ws = (char*)d_ws;
    float*  gx0p  = (float*)(ws + OFF_GX0);
    float*  xp    = (float*)(ws + OFF_XP);
    __bf16* pwhh  = (__bf16*)(ws + OFF_WHH);
    __bf16* pwih  = (__bf16*)(ws + OFF_WIH);
    __bf16* poutw = (__bf16*)(ws + OFF_OUTW);
    float*  bsum  = (float*)(ws + OFF_BSUM);

    // Phase 0: projections + weight packing (stream-ordered)
    k_xp<<<(B_SZ * HID + 255) / 256, 256, 0, stream>>>(x, lw, lb, xp);
    k_gx0<<<(B_SZ * GATES + 255) / 256, 256, 0, stream>>>(xp, w_ih, b_ih, b_hh, gx0p);
    k_bsum<<<(NLAYERS * GATES + 255) / 256, 256, 0, stream>>>(b_ih, b_hh, bsum);

    const int WELEMS = 64 * 8 * 512;  // 262144 bf16 per gate matrix
    for (int l = 0; l < NLAYERS; ++l)
        k_pack<<<(WELEMS + 255) / 256, 256, 0, stream>>>(
            w_hh + (size_t)l * GATES * HID, pwhh + (size_t)l * WELEMS, GATES, WELEMS);
    for (int l = 1; l < NLAYERS; ++l)
        k_pack<<<(WELEMS + 255) / 256, 256, 0, stream>>>(
            w_ih + (size_t)l * GATES * HID, pwih + (size_t)(l - 1) * WELEMS, GATES, WELEMS);
    const int OELEMS = 6 * 8 * 512;   // 24576 (88 rows padded to 96)
    k_pack<<<(OELEMS + 255) / 256, 256, 0, stream>>>(ow, poutw, OUTN, OELEMS);

    // Phase 1: persistent WMMA LSTM over all 64 timesteps
    k_lstm<<<B_SZ / 16, 256, 0, stream>>>(gx0p, pwhh, pwih, poutw, bsum, ob, out);
}
